// Transformer_15693810499867
// MI455X (gfx1250) — matrix operand order repaired and checked by reading, not compiled, not run on hardware
//
#include <hip/hip_runtime.h>
#include <hip/hip_bf16.h>
#include <math.h>

// ---------------------------------------------------------------------------
// Transformer forward (encoder+decoder, L=2) for MI455X / gfx1250.
// All GEMMs on the WMMA pipe: f32 -> f16 staged through LDS (vectorized,
// no bounds checks, register-double-buffered), f32 accumulate via
// v_wmma_f32_16x16x32_f16. Softmax / LayerNorm / embed are f32 VALU.
// ---------------------------------------------------------------------------

typedef __attribute__((ext_vector_type(16))) _Float16 v16h;
typedef __attribute__((ext_vector_type(8)))  _Float16 v8h;
typedef __attribute__((ext_vector_type(4)))  _Float16 v4h;
typedef __attribute__((ext_vector_type(8)))  float    v8f;
typedef __attribute__((ext_vector_type(4)))  float    v4f;

#define TB  4
#define TS  1024
#define TD  512
#define TH  8
#define TL  2
#define TDFF 2048
#define TDEPTH 64
#define TM  (TB * TS)          // 4096 activation rows

#define BM 128
#define BK 32
#define LDSPAD 8               // 8 halfs = 16B -> 80B row stride stays 16B aligned
#define LDK (BK + LDSPAD)

// ---------------------------------------------------------------------------
// Strided-batched GEMM:  C[b,h] = relu?( alpha * A[b,h] @ B[b,h] + bias )
// All dims must be tile-divisible: M%128==0, N%BNT==0, K%32==0.
//   A: [M,K] lda;  B: [K,N] ldb (TRANSB: stored [N,K]);  C: [M,N] ldc
// blockIdx.z -> (b = z/nH, h = z%nH)
// ---------------------------------------------------------------------------
template<int BNT, bool TRANSB>
__global__ __launch_bounds__(256)
void k_gemm_wmma(const float* __restrict__ A, const float* __restrict__ Bm,
                 const float* __restrict__ bias, float* __restrict__ C,
                 int K, int lda, int ldb, int ldc,
                 long long sAb, long long sAh, long long sBb, long long sBh,
                 long long sCb, long long sCh,
                 int nH, float alpha, int relu)
{
    constexpr int WNF = BNT / 32;         // WMMA frags per wave along N (4 or 2)
    constexpr int BQ  = BNT / 32;         // B-tile float4 loads per thread (4 or 2)
    constexpr int NQ  = BNT / 4;          // quads per k-row of the B tile

    __shared__ _Float16 sA[BM][LDK];
    __shared__ _Float16 sB[BNT][LDK];     // N-major: sB[n][k]

    const int z  = blockIdx.z;
    const int bb = z / nH;
    const int hh = z % nH;
    A  += (size_t)bb * sAb + (size_t)hh * sAh;
    Bm += (size_t)bb * sBb + (size_t)hh * sBh;
    C  += (size_t)bb * sCb + (size_t)hh * sCh;

    const int tid  = threadIdx.x;
    const int lane = tid & 31;
    const int wave = tid >> 5;
    const int wm   = wave & 3;            // 4 waves along M
    const int wn   = wave >> 2;           // 2 waves along N
    const int blockM = blockIdx.y * BM;
    const int blockN = blockIdx.x * BNT;

    const int row = lane & 15;
    const int hi  = lane >> 4;

    v8f acc[2][WNF];
#pragma unroll
    for (int i = 0; i < 2; ++i)
#pragma unroll
        for (int j = 0; j < WNF; ++j)
#pragma unroll
            for (int e = 0; e < 8; ++e) acc[i][j][e] = 0.f;

    v4f ra[4], rb[BQ];

    // ---- register prefetch of one k-tile (unconditional, vectorized) ----
    auto prefetch = [&](int k0) {
#pragma unroll
        for (int t = 0; t < 4; ++t) {                 // A: 1024 quads
            int qi = t * 256 + tid;
            int kq = qi & 7, m = qi >> 3;             // 8 quads per 32-wide row
            ra[t] = *(const v4f*)&A[(size_t)(blockM + m) * lda + k0 + 4 * kq];
        }
        if (TRANSB) {
#pragma unroll
            for (int t = 0; t < BQ; ++t) {            // B stored [N,K]
                int qi = t * 256 + tid;
                int kq = qi & 7, n = qi >> 3;
                rb[t] = *(const v4f*)&Bm[(size_t)(blockN + n) * ldb + k0 + 4 * kq];
            }
        } else {
#pragma unroll
            for (int t = 0; t < BQ; ++t) {            // B stored [K,N]: vector along N
                int qi = t * 256 + tid;
                int nq = qi & (NQ - 1), kk = qi / NQ;
                rb[t] = *(const v4f*)&Bm[(size_t)(k0 + kk) * ldb + blockN + 4 * nq];
            }
        }
    };

    auto store_lds = [&]() {
#pragma unroll
        for (int t = 0; t < 4; ++t) {
            int qi = t * 256 + tid;
            int kq = qi & 7, m = qi >> 3;
            v4h h;
#pragma unroll
            for (int e = 0; e < 4; ++e) h[e] = (_Float16)ra[t][e];
            *(v4h*)&sA[m][4 * kq] = h;
        }
        if (TRANSB) {
#pragma unroll
            for (int t = 0; t < BQ; ++t) {
                int qi = t * 256 + tid;
                int kq = qi & 7, n = qi >> 3;
                v4h h;
#pragma unroll
                for (int e = 0; e < 4; ++e) h[e] = (_Float16)rb[t][e];
                *(v4h*)&sB[n][4 * kq] = h;
            }
        } else {
#pragma unroll
            for (int t = 0; t < BQ; ++t) {
                int qi = t * 256 + tid;
                int nq = qi & (NQ - 1), kk = qi / NQ;
#pragma unroll
                for (int e = 0; e < 4; ++e) sB[4 * nq + e][kk] = (_Float16)rb[t][e];
            }
        }
    };

    prefetch(0);
    for (int k0 = 0; k0 < K; k0 += BK) {
        store_lds();
        __syncthreads();
        if (k0 + BK < K) prefetch(k0 + BK);   // overlap global loads with WMMA

        // A 16x32 f16 frag: lane row=lane&15; K groups {8*hi..+7},{16+8*hi..+7}
        v16h af[2];
#pragma unroll
        for (int i = 0; i < 2; ++i) {
            int m0 = wm * 32 + i * 16;
            v8h lo = *(const v8h*)&sA[m0 + row][8 * hi];
            v8h up = *(const v8h*)&sA[m0 + row][16 + 8 * hi];
#pragma unroll
            for (int e = 0; e < 8; ++e) { af[i][e] = lo[e]; af[i][8 + e] = up[e]; }
        }
        // B 32x16 f16 frag: lane col=lane&15; K groups {8*hi..+7},{16+8*hi..+7}
        v16h bf[WNF];
#pragma unroll
        for (int j = 0; j < WNF; ++j) {
            int n0 = wn * (WNF * 16) + j * 16;
            v8h blo = *(const v8h*)&sB[n0 + row][8 * hi];
            v8h bup = *(const v8h*)&sB[n0 + row][16 + 8 * hi];
#pragma unroll
            for (int e = 0; e < 8; ++e) { bf[j][e] = blo[e]; bf[j][8 + e] = bup[e]; }
        }
#pragma unroll
        for (int i = 0; i < 2; ++i)
#pragma unroll
            for (int j = 0; j < WNF; ++j)
                acc[i][j] = __builtin_amdgcn_wmma_f32_16x16x32_f16(
                    false, af[i], false, bf[j], (short)0, acc[i][j], false, false);
        __syncthreads();
    }

    // ---- epilogue (unconditional): C layout VGPR v -> M=v+8*hi, N=lane&15 ----
#pragma unroll
    for (int i = 0; i < 2; ++i) {
#pragma unroll
        for (int j = 0; j < WNF; ++j) {
            int gm0 = blockM + wm * 32 + i * 16 + 8 * hi;
            int gn  = blockN + wn * (WNF * 16) + j * 16 + row;
            float bv = bias ? bias[gn] : 0.f;
#pragma unroll
            for (int v = 0; v < 8; ++v) {
                float r = alpha * acc[i][j][v] + bv;
                if (relu && r < 0.f) r = 0.f;
                C[(size_t)(gm0 + v) * ldc + gn] = r;
            }
        }
    }
}

// ---------------------------------------------------------------------------
// Row softmax over logits[B*H*Sq, Sk] with optional causal + pad mask.
// ---------------------------------------------------------------------------
__global__ __launch_bounds__(256)
void k_softmax(float* __restrict__ logits, const float* __restrict__ padmask,
               int causal, int Sk, int nH)
{
    __shared__ float red[256];
    const int r   = blockIdx.x;          // row over B*H*Sq
    const int i   = r % Sk;              // Sq == Sk == S
    const int bh  = r / Sk;
    const int b   = bh / nH;
    float* row = logits + (size_t)r * Sk;
    const int tid = threadIdx.x;

    float mx = -3.0e38f;
    for (int j = tid; j < Sk; j += 256) {
        float v = row[j];
        if (padmask) v += padmask[(size_t)b * Sk + j] * -1e9f;
        if (causal && j > i) v += -1e9f;
        row[j] = v;
        mx = fmaxf(mx, v);
    }
    red[tid] = mx; __syncthreads();
    for (int o = 128; o > 0; o >>= 1) { if (tid < o) red[tid] = fmaxf(red[tid], red[tid + o]); __syncthreads(); }
    mx = red[0]; __syncthreads();

    float s = 0.f;
    for (int j = tid; j < Sk; j += 256) { float e = __expf(row[j] - mx); row[j] = e; s += e; }
    red[tid] = s; __syncthreads();
    for (int o = 128; o > 0; o >>= 1) { if (tid < o) red[tid] += red[tid + o]; __syncthreads(); }
    float inv = 1.f / red[0];
    for (int j = tid; j < Sk; j += 256) row[j] *= inv;
}

// ---------------------------------------------------------------------------
// out = LayerNorm(a + b) * g + be   (one block per row of width D)
// ---------------------------------------------------------------------------
__global__ __launch_bounds__(256)
void k_add_ln(const float* __restrict__ a, const float* __restrict__ b,
              const float* __restrict__ g, const float* __restrict__ be,
              float* __restrict__ out, int D)
{
    __shared__ float red[256];
    const int r = blockIdx.x;
    const float* ar = a + (size_t)r * D;
    const float* br = b + (size_t)r * D;
    float* orow = out + (size_t)r * D;
    const int tid = threadIdx.x;

    float s = 0.f;
    for (int d = tid; d < D; d += 256) s += ar[d] + br[d];
    red[tid] = s; __syncthreads();
    for (int o = 128; o > 0; o >>= 1) { if (tid < o) red[tid] += red[tid + o]; __syncthreads(); }
    float mean = red[0] / D; __syncthreads();

    float v = 0.f;
    for (int d = tid; d < D; d += 256) { float x = ar[d] + br[d] - mean; v += x * x; }
    red[tid] = v; __syncthreads();
    for (int o = 128; o > 0; o >>= 1) { if (tid < o) red[tid] += red[tid + o]; __syncthreads(); }
    float rstd = rsqrtf(red[0] / D + 1e-6f);

    for (int d = tid; d < D; d += 256) {
        float x = ar[d] + br[d];
        orow[d] = (x - mean) * rstd * g[d] + be[d];
    }
}

// ---------------------------------------------------------------------------
// y = emb[tar] * sqrt(D)
// ---------------------------------------------------------------------------
__global__ void k_embed(const int* __restrict__ tar, const float* __restrict__ emb,
                        float* __restrict__ out, int D, float scale, int total)
{
    int idx = blockIdx.x * blockDim.x + threadIdx.x;
    if (idx >= total) return;
    int r = idx / D, d = idx % D;
    out[idx] = emb[(size_t)tar[r] * D + d] * scale;
}

// ---------------------------------------------------------------------------
// Input order: top-level dict in insertion order; nested param dicts
// flattened with sorted keys (jax pytree convention).
// ---------------------------------------------------------------------------
enum {
    IN_XENC = 0, IN_TAR, IN_PADMASK, IN_EMB,
    E_B1, E_B2, E_BE1, E_BE2, E_BK, E_BO, E_BQ, E_BV, E_G1, E_G2,
    E_W1, E_W2, E_WK, E_WO, E_WQ, E_WV,
    D_B1, D_B2, D_BE1, D_BE2, D_BE3, D_BK1, D_BK2, D_BO1, D_BO2,
    D_BQ1, D_BQ2, D_BV1, D_BV2, D_G1, D_G2, D_G3, D_W1, D_W2,
    D_WK1, D_WK2, D_WO1, D_WO2, D_WQ1, D_WQ2, D_WV1, D_WV2
};

extern "C" void kernel_launch(void* const* d_in, const int* in_sizes, int n_in,
                              void* d_out, int out_size, void* d_ws, size_t ws_size,
                              hipStream_t stream)
{
    (void)in_sizes; (void)n_in; (void)out_size; (void)ws_size;
    const int M = TM, D = TD, S = TS, Hh = TH, DFF = TDFF, DEP = TDEPTH;

    const float* x_enc = (const float*)d_in[IN_XENC];
    const int*   tar   = (const int*)d_in[IN_TAR];
    const float* pad   = (const float*)d_in[IN_PADMASK];
    const float* emb   = (const float*)d_in[IN_EMB];
    auto F = [&](int i) { return (const float*)d_in[i]; };

    // -------- workspace carve (floats) --------
    float* ws = (float*)d_ws;
    const size_t MD = (size_t)M * D;
    float* qb     = ws;                 // [M,D]
    float* kb     = qb + MD;
    float* vb     = kb + MD;
    float* t1     = vb + MD;            // attn context
    float* t2     = t1 + MD;            // proj / ffn-out
    float* xA     = t2 + MD;
    float* xB     = xA + MD;
    float* encOut = xB + MD;
    float* yb     = encOut + MD;
    float* yb2    = yb + MD;
    float* o1     = yb2 + MD;
    float* o2     = o1 + MD;
    float* ffh    = qb;                 // alias: [M,DFF] == 4*MD (q/k/v/t1 dead in FFN)
    float* logits = o2 + MD;            // [B*H*S*S]

    auto gemm = [&](const float* A, const float* Bm, const float* bias, float* Cp,
                    int M_, int N_, int K_, int lda, int ldb, int ldc,
                    long long sAb, long long sAh, long long sBb, long long sBh,
                    long long sCb, long long sCh,
                    int nH, int nZ, int transB, float alpha, int relu) {
        if (N_ % 128 == 0) {
            dim3 grid(N_ / 128, M_ / BM, nZ);
            if (transB)
                k_gemm_wmma<128, true><<<grid, 256, 0, stream>>>(A, Bm, bias, Cp,
                    K_, lda, ldb, ldc, sAb, sAh, sBb, sBh, sCb, sCh, nH, alpha, relu);
            else
                k_gemm_wmma<128, false><<<grid, 256, 0, stream>>>(A, Bm, bias, Cp,
                    K_, lda, ldb, ldc, sAb, sAh, sBb, sBh, sCb, sCh, nH, alpha, relu);
        } else {                        // N == 64 (probs @ V)
            dim3 grid(N_ / 64, M_ / BM, nZ);
            k_gemm_wmma<64, false><<<grid, 256, 0, stream>>>(A, Bm, bias, Cp,
                K_, lda, ldb, ldc, sAb, sAh, sBb, sBh, sCb, sCh, nH, alpha, relu);
        }
    };

    const long long sBH = (long long)S * D;        // batch(b) stride in [B*S,D]
    const long long sLb = (long long)Hh * S * S;   // batch(b) stride in logits
    const long long sLh = (long long)S * S;
    const float iscale = 0.125f;                   // 1/sqrt(DEPTH)

    auto attention = [&](const float* xq, const float* xkv,
                         const float* wq, const float* bq,
                         const float* wk, const float* bk,
                         const float* wv, const float* bv,
                         const float* wo, const float* bo,
                         const float* padmask, int causal, float* outp) {
        gemm(xq,  wq, bq, qb, M, D, D, D, D, D, 0,0,0,0,0,0, 1, 1, 0, 1.f, 0);
        gemm(xkv, wk, bk, kb, M, D, D, D, D, D, 0,0,0,0,0,0, 1, 1, 0, 1.f, 0);
        gemm(xkv, wv, bv, vb, M, D, D, D, D, D, 0,0,0,0,0,0, 1, 1, 0, 1.f, 0);
        // logits[b,h] = Q[b,h] @ K[b,h]^T * iscale   (M=S, N=S, K=64)
        gemm(qb, kb, nullptr, logits, S, S, DEP, D, D, S,
             sBH, DEP, sBH, DEP, sLb, sLh, Hh, TB * Hh, 1, iscale, 0);
        k_softmax<<<TB * Hh * S, 256, 0, stream>>>(logits, padmask, causal, S, Hh);
        // ctx[b,h] = probs @ V   (M=S, N=64, K=S)
        gemm(logits, vb, nullptr, t1, S, DEP, S, S, D, D,
             sLb, sLh, sBH, DEP, sBH, DEP, Hh, TB * Hh, 0, 1.f, 0);
        gemm(t1, wo, bo, outp, M, D, D, D, D, D, 0,0,0,0,0,0, 1, 1, 0, 1.f, 0);
    };

    // ================= Encoder =================
    const float* xin = x_enc;
    for (int l = 0; l < TL; ++l) {
        const size_t wDD = (size_t)l * D * D, bD = (size_t)l * D;
        attention(xin, xin,
                  F(E_WQ) + wDD, F(E_BQ) + bD, F(E_WK) + wDD, F(E_BK) + bD,
                  F(E_WV) + wDD, F(E_BV) + bD, F(E_WO) + wDD, F(E_BO) + bD,
                  pad, 0, t2);
        k_add_ln<<<M, 256, 0, stream>>>(xin, t2, F(E_G1) + bD, F(E_BE1) + bD, xA, D);
        gemm(xA, F(E_W1) + (size_t)l * D * DFF, F(E_B1) + (size_t)l * DFF, ffh,
             M, DFF, D, D, DFF, DFF, 0,0,0,0,0,0, 1, 1, 0, 1.f, 1);
        gemm(ffh, F(E_W2) + (size_t)l * DFF * D, F(E_B2) + bD, t2,
             M, D, DFF, DFF, D, D, 0,0,0,0,0,0, 1, 1, 0, 1.f, 0);
        float* xout = (l == TL - 1) ? encOut : xB;
        k_add_ln<<<M, 256, 0, stream>>>(xA, t2, F(E_G2) + bD, F(E_BE2) + bD, xout, D);
        xin = xout;
    }

    // ================= Decoder =================
    k_embed<<<((int)MD + 255) / 256, 256, 0, stream>>>(tar, emb, yb, D,
                                                       22.62741699796952f, (int)MD);
    const float* yin = yb;
    for (int l = 0; l < TL; ++l) {
        const size_t wDD = (size_t)l * D * D, bD = (size_t)l * D;
        attention(yin, yin,
                  F(D_WQ1) + wDD, F(D_BQ1) + bD, F(D_WK1) + wDD, F(D_BK1) + bD,
                  F(D_WV1) + wDD, F(D_BV1) + bD, F(D_WO1) + wDD, F(D_BO1) + bD,
                  nullptr, 1, t2);
        k_add_ln<<<M, 256, 0, stream>>>(yin, t2, F(D_G1) + bD, F(D_BE1) + bD, o1, D);
        attention(o1, encOut,
                  F(D_WQ2) + wDD, F(D_BQ2) + bD, F(D_WK2) + wDD, F(D_BK2) + bD,
                  F(D_WV2) + wDD, F(D_BV2) + bD, F(D_WO2) + wDD, F(D_BO2) + bD,
                  pad, 0, t2);
        k_add_ln<<<M, 256, 0, stream>>>(o1, t2, F(D_G2) + bD, F(D_BE2) + bD, o2, D);
        gemm(o2, F(D_W1) + (size_t)l * D * DFF, F(D_B1) + (size_t)l * DFF, ffh,
             M, DFF, D, D, DFF, DFF, 0,0,0,0,0,0, 1, 1, 0, 1.f, 1);
        gemm(ffh, F(D_W2) + (size_t)l * DFF * D, F(D_B2) + bD, t2,
             M, D, DFF, DFF, D, D, 0,0,0,0,0,0, 1, 1, 0, 1.f, 0);
        float* yout = (l == TL - 1) ? (float*)d_out : yb2;
        k_add_ln<<<M, 256, 0, stream>>>(o2, t2, F(D_G3) + bD, F(D_BE3) + bD, yout, D);
        yin = yout;
    }
}
